// VQVAE_45045617001179
// MI455X (gfx1250) — compile-verified, matrix operand order checked by reading, not run on hardware
//
#include <hip/hip_runtime.h>
#include <hip/hip_bf16.h>

// ---------------------------------------------------------------------------
// CDNA5 (gfx1250) VQ-VAE forward.  All heavy math runs through
// v_wmma_f32_16x16x32_bf16 (wave32, 16x16x32 tiles).
// GEMM: 256x128 block tile, 8 waves (4Mx2N), 64x64 per wave (4x4 fragments),
// double-buffered LDS (one barrier per K-step).
// ---------------------------------------------------------------------------

typedef __attribute__((ext_vector_type(16))) __bf16 v16bf;
typedef __attribute__((ext_vector_type(8)))  float  v8f;

union Frag16 { v16bf v; uint4 q[2]; };

// ---------------------------------------------------------------------------
// im2col: NCHW fp32 -> (Mpad x Kpad) bf16 row-major patch matrix (zero padded)
// rows m = (b, oy, ox), cols k = c*kh*kw + ky*kw + kx   (matches OIHW weights)
// ---------------------------------------------------------------------------
__global__ void k_im2col(const float* __restrict__ in, __bf16* __restrict__ out,
                         int B, int C, int H, int W, int Ho, int Wo,
                         int kh, int kw, int stride, int pad,
                         int Kreal, int Kpad, int Mreal, unsigned long long total)
{
    unsigned long long i = (unsigned long long)blockIdx.x * blockDim.x + threadIdx.x;
    if (i >= total) return;
    int k = (int)(i % Kpad);
    long long m = (long long)(i / Kpad);
    float v = 0.f;
    if (m < Mreal && k < Kreal) {
        int khw = kh * kw;
        int c = k / khw, p = k % khw;
        int ky = p / kw, kx = p % kw;
        int hw = Ho * Wo;
        int b = (int)(m / hw), r = (int)(m % hw);
        int oy = r / Wo, ox = r % Wo;
        int iy = oy * stride - pad + ky;
        int ix = ox * stride - pad + kx;
        if ((unsigned)iy < (unsigned)H && (unsigned)ix < (unsigned)W)
            v = in[(((size_t)b * C + c) * H + iy) * W + ix];
    }
    out[i] = (__bf16)v;
}

// fp32 (Nreal x Kreal) row-major -> bf16 (Npad x Kpad), zero padded
__global__ void k_wcast(const float* __restrict__ w, __bf16* __restrict__ out,
                        int Nreal, int Kreal, int Kpad, unsigned long long total)
{
    unsigned long long i = (unsigned long long)blockIdx.x * blockDim.x + threadIdx.x;
    if (i >= total) return;
    int k = (int)(i % Kpad);
    int n = (int)(i / Kpad);
    float v = (n < Nreal && k < Kreal) ? w[(size_t)n * Kreal + k] : 0.f;
    out[i] = (__bf16)v;
}

// ConvT weight (I, O, khw) -> bf16 B' (Npad x Kpad), B'[o*khw+p, i] = w[i,o,p]
__global__ void k_wcastT(const float* __restrict__ w, __bf16* __restrict__ out,
                         int I, int O, int khw, int Kpad, unsigned long long total)
{
    unsigned long long i = (unsigned long long)blockIdx.x * blockDim.x + threadIdx.x;
    if (i >= total) return;
    int k = (int)(i % Kpad);
    int n = (int)(i / Kpad);
    float v = 0.f;
    if (n < O * khw && k < I) {
        int o = n / khw, p = n % khw;
        v = w[((size_t)k * O + o) * khw + p];
    }
    out[i] = (__bf16)v;
}

// ---------------------------------------------------------------------------
// WMMA bf16 GEMM:  C[M,N] = A[M,K] * B^T[N,K]
// block = 256 threads = 8 wave32 waves; block tile 256(M) x 128(N), K-step 32.
// wave grid 4(M) x 2(N); each wave owns 4x4 16x16 accumulators (64x64).
// A-fragment layout per CDNA5 ISA: lane<16 holds row m=lane, K {0-7,16-23};
// lane>=16 holds row m=lane-16, K {8-15,24-31}.  B loaded identically from
// the row-major N x K (= B^T) operand.
// LDS double buffered: 2 x (16KB A + 8KB B) = 48KB; one barrier per K-step.
// mode 0: C row-major Mreal x Nreal.  mode 1: NCHW scatter + bias + relu.
// ---------------------------------------------------------------------------
__global__ __launch_bounds__(256)
void k_gemm_bf16(const __bf16* __restrict__ A, const __bf16* __restrict__ B,
                 float* __restrict__ C, int K,
                 int mode, int Mreal, int Nreal,
                 const float* __restrict__ bias, int relu, int HW)
{
    __shared__ __bf16 As[2][256 * 32];
    __shared__ __bf16 Bs[2][128 * 32];

    const int tid  = threadIdx.x;
    const int lane = tid & 31;
    const int wave = tid >> 5;
    const int wm   = wave & 3;   // 0..3 along M (64 rows each)
    const int wn   = wave >> 2;  // 0..1 along N (64 cols each)
    const int mBase = blockIdx.y * 256;
    const int nBase = blockIdx.x * 128;

    v8f acc[4][4];
#pragma unroll
    for (int i = 0; i < 4; ++i)
#pragma unroll
        for (int j = 0; j < 4; ++j)
#pragma unroll
            for (int e = 0; e < 8; ++e) acc[i][j][e] = 0.f;

    // cooperative loaders:
    //  A: thread t -> full 64B row (mBase+t), 4 x uint4
    //  B: thread t -> row (nBase + t/2), 32B half (t&1), 2 x uint4
    const uint4* ag = reinterpret_cast<const uint4*>(A + (size_t)(mBase + tid) * K);
    const int brow = tid >> 1, bhalf = tid & 1;
    const uint4* bg = reinterpret_cast<const uint4*>(B + (size_t)(nBase + brow) * K) + bhalf * 2;

    uint4 a0, a1, a2, a3, b0, b1;
    a0 = ag[0]; a1 = ag[1]; a2 = ag[2]; a3 = ag[3];
    b0 = bg[0]; b1 = bg[1];
    {
        uint4* aw = reinterpret_cast<uint4*>(&As[0][tid * 32]);
        aw[0] = a0; aw[1] = a1; aw[2] = a2; aw[3] = a3;
        uint4* bw = reinterpret_cast<uint4*>(&Bs[0][brow * 32]) + bhalf * 2;
        bw[0] = b0; bw[1] = b1;
    }
    __syncthreads();

    const int lr = lane & 15;   // row within 16x16 sub-tile
    const int lh = lane >> 4;   // K-half selector
    const int nIter = K >> 5;

    for (int it = 0; it < nIter; ++it) {
        const int buf = it & 1;
        const bool more = (it + 1) < nIter;
        if (more) {
            const uint4* ap = ag + ((it + 1) << 2);   // 32 bf16 = 4 uint4 per step
            a0 = ap[0]; a1 = ap[1]; a2 = ap[2]; a3 = ap[3];
            const uint4* bp = bg + ((it + 1) << 2);
            b0 = bp[0]; b1 = bp[1];
        }

        Frag16 bfr[4];
#pragma unroll
        for (int ni = 0; ni < 4; ++ni) {
            const uint4* p = reinterpret_cast<const uint4*>(&Bs[buf][(wn * 64 + ni * 16 + lr) * 32]);
            bfr[ni].q[0] = p[lh];
            bfr[ni].q[1] = p[lh + 2];
        }
#pragma unroll
        for (int mi = 0; mi < 4; ++mi) {
            Frag16 af;
            const uint4* p = reinterpret_cast<const uint4*>(&As[buf][(wm * 64 + mi * 16 + lr) * 32]);
            af.q[0] = p[lh];
            af.q[1] = p[lh + 2];
#pragma unroll
            for (int ni = 0; ni < 4; ++ni)
                acc[mi][ni] = __builtin_amdgcn_wmma_f32_16x16x32_bf16(
                    false, af.v, false, bfr[ni].v,
                    (short)0, acc[mi][ni], false, false);
        }

        if (more) {
            // write next tile into the other buffer; reads of that buffer
            // (previous iteration) are ordered by the previous barrier.
            uint4* aw = reinterpret_cast<uint4*>(&As[buf ^ 1][tid * 32]);
            aw[0] = a0; aw[1] = a1; aw[2] = a2; aw[3] = a3;
            uint4* bw = reinterpret_cast<uint4*>(&Bs[buf ^ 1][brow * 32]) + bhalf * 2;
            bw[0] = b0; bw[1] = b1;
            __syncthreads();
        }
    }

    // epilogue: C/D layout -> element (M = e + 8*lh, N = lr) per VGPR e
#pragma unroll
    for (int mi = 0; mi < 4; ++mi)
#pragma unroll
        for (int ni = 0; ni < 4; ++ni)
#pragma unroll
            for (int e = 0; e < 8; ++e) {
                int m = mBase + wm * 64 + mi * 16 + lh * 8 + e;
                int n = nBase + wn * 64 + ni * 16 + lr;
                if (m < Mreal && n < Nreal) {
                    float v = acc[mi][ni][e];
                    if (mode == 1) {
                        if (bias) v += bias[n];
                        if (relu) v = fmaxf(v, 0.f);
                        int b = m / HW, hw = m - b * HW;
                        C[((size_t)b * Nreal + n) * HW + hw] = v;
                    } else {
                        C[(size_t)m * Nreal + n] = v;
                    }
                }
            }
}

// ---------------------------------------------------------------------------
// BatchNorm (training mode): per-channel mean/var over (N,H,W), fold into
// scale/shift, then fused apply (+optional residual, +optional relu).
// ---------------------------------------------------------------------------
__global__ void k_bn_stats(const float* __restrict__ x, const float* __restrict__ g,
                           const float* __restrict__ bt, float* __restrict__ scale,
                           float* __restrict__ shift, int B, int C, int HW)
{
    int c = blockIdx.x;
    __shared__ float s1[256], s2[256];
    float a = 0.f, b2 = 0.f;
    int n = B * HW;
    for (int i = threadIdx.x; i < n; i += 256) {
        int bb = i / HW, hw = i % HW;
        float v = x[((size_t)bb * C + c) * HW + hw];
        a += v; b2 += v * v;
    }
    s1[threadIdx.x] = a; s2[threadIdx.x] = b2;
    __syncthreads();
    for (int st = 128; st > 0; st >>= 1) {
        if (threadIdx.x < st) {
            s1[threadIdx.x] += s1[threadIdx.x + st];
            s2[threadIdx.x] += s2[threadIdx.x + st];
        }
        __syncthreads();
    }
    if (threadIdx.x == 0) {
        float mean = s1[0] / n;
        float var  = s2[0] / n - mean * mean;
        float sc   = g[c] * rsqrtf(var + 1e-5f);
        scale[c] = sc;
        shift[c] = bt[c] - mean * sc;
    }
}

__global__ void k_bn_apply(const float* __restrict__ x, const float* __restrict__ res,
                           float* __restrict__ out,
                           const float* __restrict__ scale, const float* __restrict__ shift,
                           int C, int HW, int relu, unsigned long long total)
{
    unsigned long long i = (unsigned long long)blockIdx.x * blockDim.x + threadIdx.x;
    if (i >= total) return;
    int c = (int)((i / HW) % C);
    float v = x[i] * scale[c] + shift[c];
    if (res)  v += res[i];
    if (relu) v = fmaxf(v, 0.f);
    out[i] = v;
}

// ---------------------------------------------------------------------------
// Vector quantization helpers
// ---------------------------------------------------------------------------
__global__ void k_rownorm(const float* __restrict__ emb, float* __restrict__ en, int D)
{
    int j = blockIdx.x;
    __shared__ float s[256];
    float a = 0.f;
    for (int k = threadIdx.x; k < D; k += 256) {
        float v = emb[(size_t)j * D + k];
        a += v * v;
    }
    s[threadIdx.x] = a;
    __syncthreads();
    for (int st = 128; st > 0; st >>= 1) {
        if (threadIdx.x < st) s[threadIdx.x] += s[threadIdx.x + st];
        __syncthreads();
    }
    if (threadIdx.x == 0) en[j] = s[0];
}

__global__ void k_argmin(const float* __restrict__ dots, const float* __restrict__ en,
                         int* __restrict__ idx, int Ncode)
{
    int m = blockIdx.x;
    __shared__ float sv[256];
    __shared__ int   si[256];
    float best = __builtin_inff();
    int bj = 0;
    for (int j = threadIdx.x; j < Ncode; j += 256) {
        float s = en[j] - 2.f * dots[(size_t)m * Ncode + j];
        if (s < best || (s == best && j < bj)) { best = s; bj = j; }
    }
    sv[threadIdx.x] = best; si[threadIdx.x] = bj;
    __syncthreads();
    for (int st = 128; st > 0; st >>= 1) {
        if (threadIdx.x < st) {
            float s2 = sv[threadIdx.x + st]; int j2 = si[threadIdx.x + st];
            if (s2 < sv[threadIdx.x] || (s2 == sv[threadIdx.x] && j2 < si[threadIdx.x])) {
                sv[threadIdx.x] = s2; si[threadIdx.x] = j2;
            }
        }
        __syncthreads();
    }
    if (threadIdx.x == 0) idx[m] = si[0];
}

// faithful to reference: flat gather + direct reshape to NCHW
__global__ void k_gather(const float* __restrict__ emb, const int* __restrict__ idx,
                         float* __restrict__ out, int D, unsigned long long total)
{
    unsigned long long i = (unsigned long long)blockIdx.x * blockDim.x + threadIdx.x;
    if (i >= total) return;
    unsigned long long row = i / (unsigned)D;
    int col = (int)(i % (unsigned)D);
    out[i] = emb[(size_t)idx[row] * D + col];
}

// ---------------------------------------------------------------------------
// ConvT col2im-gather: out[b,o,oy,ox] = bias[o] + sum_valid G[m(iy,ix), o*khw+p]
// deterministic (no atomics).  G row stride = Nreal = O*kh*kw.
// ---------------------------------------------------------------------------
__global__ void k_colT_gather(const float* __restrict__ G, const float* __restrict__ bias,
                              float* __restrict__ out,
                              int O, int kh, int kw, int IH, int IW, int OH, int OW,
                              int stride, int pad, int relu, unsigned long long total)
{
    unsigned long long i = (unsigned long long)blockIdx.x * blockDim.x + threadIdx.x;
    if (i >= total) return;
    int ox = (int)(i % OW);
    unsigned long long t = i / OW;
    int oy = (int)(t % OH); t /= OH;
    int o  = (int)(t % O);
    int b  = (int)(t / O);
    int khw = kh * kw;
    float s = bias[o];
    for (int ky = 0; ky < kh; ++ky) {
        int ty = oy + pad - ky;
        if (ty < 0 || (ty % stride) != 0) continue;
        int iy = ty / stride;
        if (iy >= IH) continue;
        for (int kx = 0; kx < kw; ++kx) {
            int tx = ox + pad - kx;
            if (tx < 0 || (tx % stride) != 0) continue;
            int ix = tx / stride;
            if (ix >= IW) continue;
            int m = (b * IH + iy) * IW + ix;
            s += G[(size_t)m * (O * khw) + o * khw + ky * kw + kx];
        }
    }
    if (relu) s = fmaxf(s, 0.f);
    out[i] = s;
}

// ---------------------------------------------------------------------------
// small direct kernels for the cheap edge layers
// ---------------------------------------------------------------------------
__global__ void k_conv_e1(const float* __restrict__ x, const float* __restrict__ w,
                          const float* __restrict__ bias, float* __restrict__ out)
{
    // in (8,3,128,128) -> out (8,64,63,63), k5 s2 p1, +bias, relu
    int i = blockIdx.x * blockDim.x + threadIdx.x;
    if (i >= 8 * 64 * 63 * 63) return;
    int ox = i % 63; int t = i / 63;
    int oy = t % 63; t /= 63;
    int o  = t % 64; int b = t / 64;
    float s = bias[o];
    for (int c = 0; c < 3; ++c)
        for (int ky = 0; ky < 5; ++ky) {
            int iy = oy * 2 - 1 + ky;
            if ((unsigned)iy >= 128u) continue;
            for (int kx = 0; kx < 5; ++kx) {
                int ix = ox * 2 - 1 + kx;
                if ((unsigned)ix >= 128u) continue;
                s += x[(((size_t)b * 3 + c) * 128 + iy) * 128 + ix] *
                     w[(((size_t)o * 3 + c) * 5 + ky) * 5 + kx];
            }
        }
    out[i] = fmaxf(s, 0.f);
}

__global__ void k_convt_d3(const float* __restrict__ h, const float* __restrict__ w,
                           const float* __restrict__ bias, float* __restrict__ out)
{
    // h (8,64,65,65), w (64,1,3,3), out (8,1,129,129), k3 s2 p1 (no relu)
    int i = blockIdx.x * blockDim.x + threadIdx.x;
    if (i >= 8 * 129 * 129) return;
    int ox = i % 129; int t = i / 129;
    int oy = t % 129; int b = t / 129;
    float s = bias[0];
    for (int ky = 0; ky < 3; ++ky) {
        int ty = oy + 1 - ky;
        if (ty < 0 || (ty & 1)) continue;
        int iy = ty >> 1;
        if (iy >= 65) continue;
        for (int kx = 0; kx < 3; ++kx) {
            int tx = ox + 1 - kx;
            if (tx < 0 || (tx & 1)) continue;
            int ix = tx >> 1;
            if (ix >= 65) continue;
            const float* hp = h + (size_t)b * 64 * 65 * 65 + (size_t)iy * 65 + ix;
            float wk0 = 0.f;
            for (int c = 0; c < 64; ++c)
                wk0 += hp[(size_t)c * 65 * 65] * w[c * 9 + ky * 3 + kx];
            s += wk0;
        }
    }
    out[i] = s;
}

// ---------------------------------------------------------------------------
// host-side orchestration
// ---------------------------------------------------------------------------
static inline unsigned gblk(unsigned long long n) { return (unsigned)((n + 255) / 256); }

extern "C" void kernel_launch(void* const* d_in, const int* in_sizes, int n_in,
                              void* d_out, int out_size, void* d_ws, size_t ws_size,
                              hipStream_t stream)
{
    (void)in_sizes; (void)n_in; (void)out_size; (void)ws_size;

    const float* x       = (const float*)d_in[0];
    const float* e1_w    = (const float*)d_in[1];  const float* e1_b = (const float*)d_in[2];
    const float* e2_w    = (const float*)d_in[3];  const float* e2_b = (const float*)d_in[4];
    const float* e3_w    = (const float*)d_in[5];  const float* e3_b = (const float*)d_in[6];
    const float* re_c1_w = (const float*)d_in[7];  const float* re_c1_b = (const float*)d_in[8];
    const float* re_g1   = (const float*)d_in[9];  const float* re_b1   = (const float*)d_in[10];
    const float* re_c2_w = (const float*)d_in[11];
    const float* re_g2   = (const float*)d_in[12]; const float* re_b2   = (const float*)d_in[13];
    const float* rd_c1_w = (const float*)d_in[14]; const float* rd_c1_b = (const float*)d_in[15];
    const float* rd_g1   = (const float*)d_in[16]; const float* rd_b1   = (const float*)d_in[17];
    const float* rd_c2_w = (const float*)d_in[18];
    const float* rd_g2   = (const float*)d_in[19]; const float* rd_b2   = (const float*)d_in[20];
    const float* d1_w    = (const float*)d_in[21]; const float* d1_b = (const float*)d_in[22];
    const float* d2_w    = (const float*)d_in[23]; const float* d2_b = (const float*)d_in[24];
    const float* d3_w    = (const float*)d_in[25]; const float* d3_b = (const float*)d_in[26];
    const float* emb     = (const float*)d_in[27];

    // outputs: x_hat (8*129*129) | z_e (8*2048*16*16) | z_q (same)
    float* outp = (float*)d_out;
    float* xhat = outp;
    float* z_e  = outp + 133128;
    float* z_q  = z_e + 4194304;

    // workspace (assumes ~260MB available; all offsets 256B aligned)
    char* ws = (char*)d_ws;
    size_t off = 0;
    auto take = [&](size_t bytes) -> char* {
        char* p = ws + off;
        off = (off + bytes + 255) & ~(size_t)255;
        return p;
    };
    float*  h1   = (float*)take((size_t)2032128 * 4);           // (8,64,63,63)
    float*  h2   = (float*)take((size_t)1048576 * 4);           // (8,128,32,32)
    float*  h3   = (float*)take((size_t)4194304 * 4);           // (8,2048,16,16), reused decoder-side
    float*  t1   = (float*)take((size_t)4194304 * 4);
    float*  t2   = (float*)take((size_t)4194304 * 4);
    __bf16* Acol = (__bf16*)take((size_t)37748736 * 2);         // max 2048 x 18432
    __bf16* Bw   = (__bf16*)take((size_t)37748736 * 2);         // max 2048 x 18432
    float*  G    = (float*)take((size_t)6553600 * 4);           // max 2048 x 3200
    float*  dots = (float*)take((size_t)2048 * 512 * 4);
    float*  en   = (float*)take((size_t)512 * 4);
    int*    vidx = (int*)take((size_t)2048 * 4);
    float*  bsc  = (float*)take((size_t)2048 * 4);
    float*  bsh  = (float*)take((size_t)2048 * 4);
    float*  d1o  = (float*)take((size_t)1115136 * 4);           // (8,128,33,33)
    float*  d2o  = (float*)take((size_t)2163200 * 4);           // (8,64,65,65)

    const dim3 B256(256);

    // ---- encoder ----
    // e1: direct conv 3->64 k5 s2 p1 + relu
    k_conv_e1<<<gblk(2032128), B256, 0, stream>>>(x, e1_w, e1_b, h1);

    // e2: 64->128 k3 s2 p1  (M=8192, N=128, K=576)
    k_im2col<<<gblk(8192ull * 576), B256, 0, stream>>>(h1, Acol, 8, 64, 63, 63, 32, 32,
                                                       3, 3, 2, 1, 576, 576, 8192, 8192ull * 576);
    k_wcast<<<gblk(128ull * 576), B256, 0, stream>>>(e2_w, Bw, 128, 576, 576, 128ull * 576);
    k_gemm_bf16<<<dim3(1, 32), B256, 0, stream>>>(Acol, Bw, h2, 576, 1, 8192, 128, e2_b, 1, 1024);

    // e3: 128->2048 k3 s2 p1  (M=2048, N=2048, K=1152)
    k_im2col<<<gblk(2048ull * 1152), B256, 0, stream>>>(h2, Acol, 8, 128, 32, 32, 16, 16,
                                                        3, 3, 2, 1, 1152, 1152, 2048, 2048ull * 1152);
    k_wcast<<<gblk(2048ull * 1152), B256, 0, stream>>>(e3_w, Bw, 2048, 1152, 1152, 2048ull * 1152);
    k_gemm_bf16<<<dim3(16, 8), B256, 0, stream>>>(Acol, Bw, h3, 1152, 1, 2048, 2048, e3_b, 1, 256);

    // ---- encoder resblock (M=2048, N=2048, K=18432) ----
    k_im2col<<<gblk(2048ull * 18432), B256, 0, stream>>>(h3, Acol, 8, 2048, 16, 16, 16, 16,
                                                         3, 3, 1, 1, 18432, 18432, 2048, 2048ull * 18432);
    k_wcast<<<gblk(2048ull * 18432), B256, 0, stream>>>(re_c1_w, Bw, 2048, 18432, 18432, 2048ull * 18432);
    k_gemm_bf16<<<dim3(16, 8), B256, 0, stream>>>(Acol, Bw, t1, 18432, 1, 2048, 2048, re_c1_b, 0, 256);
    k_bn_stats<<<2048, B256, 0, stream>>>(t1, re_g1, re_b1, bsc, bsh, 8, 2048, 256);
    k_bn_apply<<<gblk(4194304), B256, 0, stream>>>(t1, (const float*)nullptr, t1, bsc, bsh, 2048, 256, 1, 4194304);

    k_im2col<<<gblk(2048ull * 18432), B256, 0, stream>>>(t1, Acol, 8, 2048, 16, 16, 16, 16,
                                                         3, 3, 1, 1, 18432, 18432, 2048, 2048ull * 18432);
    k_wcast<<<gblk(2048ull * 18432), B256, 0, stream>>>(re_c2_w, Bw, 2048, 18432, 18432, 2048ull * 18432);
    k_gemm_bf16<<<dim3(16, 8), B256, 0, stream>>>(Acol, Bw, t2, 18432, 1, 2048, 2048, (const float*)nullptr, 0, 256);
    k_bn_stats<<<2048, B256, 0, stream>>>(t2, re_g2, re_b2, bsc, bsh, 8, 2048, 256);
    k_bn_apply<<<gblk(4194304), B256, 0, stream>>>(t2, h3, z_e, bsc, bsh, 2048, 256, 1, 4194304);

    // ---- vector quantization ----
    // flat rows (b,h,w) x C  == 1x1 im2col of z_e;  dots = flat @ emb^T via WMMA
    k_im2col<<<gblk(2048ull * 2048), B256, 0, stream>>>(z_e, Acol, 8, 2048, 16, 16, 16, 16,
                                                        1, 1, 1, 0, 2048, 2048, 2048, 2048ull * 2048);
    k_wcast<<<gblk(512ull * 2048), B256, 0, stream>>>(emb, Bw, 512, 2048, 2048, 512ull * 2048);
    k_gemm_bf16<<<dim3(4, 8), B256, 0, stream>>>(Acol, Bw, dots, 2048, 0, 2048, 512, (const float*)nullptr, 0, 0);
    k_rownorm<<<512, B256, 0, stream>>>(emb, en, 2048);
    k_argmin<<<2048, B256, 0, stream>>>(dots, en, vidx, 512);
    k_gather<<<gblk(4194304), B256, 0, stream>>>(emb, vidx, z_q, 2048, 4194304);

    // ---- decoder resblock on z_q ----
    k_im2col<<<gblk(2048ull * 18432), B256, 0, stream>>>(z_q, Acol, 8, 2048, 16, 16, 16, 16,
                                                         3, 3, 1, 1, 18432, 18432, 2048, 2048ull * 18432);
    k_wcast<<<gblk(2048ull * 18432), B256, 0, stream>>>(rd_c1_w, Bw, 2048, 18432, 18432, 2048ull * 18432);
    k_gemm_bf16<<<dim3(16, 8), B256, 0, stream>>>(Acol, Bw, t1, 18432, 1, 2048, 2048, rd_c1_b, 0, 256);
    k_bn_stats<<<2048, B256, 0, stream>>>(t1, rd_g1, rd_b1, bsc, bsh, 8, 2048, 256);
    k_bn_apply<<<gblk(4194304), B256, 0, stream>>>(t1, (const float*)nullptr, t1, bsc, bsh, 2048, 256, 1, 4194304);

    k_im2col<<<gblk(2048ull * 18432), B256, 0, stream>>>(t1, Acol, 8, 2048, 16, 16, 16, 16,
                                                         3, 3, 1, 1, 18432, 18432, 2048, 2048ull * 18432);
    k_wcast<<<gblk(2048ull * 18432), B256, 0, stream>>>(rd_c2_w, Bw, 2048, 18432, 18432, 2048ull * 18432);
    k_gemm_bf16<<<dim3(16, 8), B256, 0, stream>>>(Acol, Bw, t2, 18432, 1, 2048, 2048, (const float*)nullptr, 0, 256);
    k_bn_stats<<<2048, B256, 0, stream>>>(t2, rd_g2, rd_b2, bsc, bsh, 8, 2048, 256);
    k_bn_apply<<<gblk(4194304), B256, 0, stream>>>(t2, z_q, h3 /*h4*/, bsc, bsh, 2048, 256, 1, 4194304);

    // ---- d1: convT 2048->128 k5 s2 p1  via input-side GEMM + col2im gather ----
    // G[M=2048, N=128*25=3200] = rows(h4) @ W'(n=(o,p), k=i)^T   (K=2048)
    k_im2col<<<gblk(2048ull * 2048), B256, 0, stream>>>(h3, Acol, 8, 2048, 16, 16, 16, 16,
                                                        1, 1, 1, 0, 2048, 2048, 2048, 2048ull * 2048);
    k_wcastT<<<gblk(3200ull * 2048), B256, 0, stream>>>(d1_w, Bw, 2048, 128, 25, 2048, 3200ull * 2048);
    k_gemm_bf16<<<dim3(25, 8), B256, 0, stream>>>(Acol, Bw, G, 2048, 0, 2048, 3200, (const float*)nullptr, 0, 0);
    k_colT_gather<<<gblk(1115136), B256, 0, stream>>>(G, d1_b, d1o, 128, 5, 5, 16, 16, 33, 33, 2, 1, 1, 1115136);

    // ---- d2: convT 128->64 k3 s2 p1  (M=8712 pad 8960, N=576 pad 640, K=128) ----
    k_im2col<<<gblk(8960ull * 128), B256, 0, stream>>>(d1o, Acol, 8, 128, 33, 33, 33, 33,
                                                       1, 1, 1, 0, 128, 128, 8712, 8960ull * 128);
    k_wcastT<<<gblk(640ull * 128), B256, 0, stream>>>(d2_w, Bw, 128, 64, 9, 128, 640ull * 128);
    k_gemm_bf16<<<dim3(5, 35), B256, 0, stream>>>(Acol, Bw, G, 128, 0, 8712, 576, (const float*)nullptr, 0, 0);
    k_colT_gather<<<gblk(2163200), B256, 0, stream>>>(G, d2_b, d2o, 64, 3, 3, 33, 33, 65, 65, 2, 1, 1, 2163200);

    // ---- d3: convT 64->1 k3 s2 p1 (direct, no relu) -> x_hat ----
    k_convt_d3<<<gblk(133128), B256, 0, stream>>>(d2o, d3_w, d3_b, xhat);
}
